// RNNAutoencoder_78675210928203
// MI455X (gfx1250) — matrix operand order repaired and checked by reading, not compiled, not run on hardware
//
#include <hip/hip_runtime.h>
#include <hip/hip_bf16.h>

// ---------------------------------------------------------------------------
// LSTM autoencoder on MI455X (gfx1250), bf16 WMMA + f32 accumulate.
//   B=4096, HID=1024, EMB=Y_IN=512, T=10.
// Algebraic optimizations:
//   * encoder input GEMM hoisted out of the time loop (input constant per step)
//   * mean over T pushed through the linear p2 layer (accumulate relu(p1) only)
// GEMM: 64x32 wave tile (4x2 WMMA frags) -> 1.5 b128 loads per WMMA,
// 8 waves/block as 2(M)x4(N) -> 128x128 block tile.
// ---------------------------------------------------------------------------

typedef __bf16 bf16_t;
typedef __attribute__((ext_vector_type(16))) __bf16 v16bf;
typedef __attribute__((ext_vector_type(8)))  float  v8f;

#define B_ROWS 4096
#define YIN    512
#define HIDN   1024
#define EMBD   512
#define G4     4096   // 4*HID
#define TSTEPS 10

__device__ __forceinline__ bf16_t f2bf(float f) {
  // round-to-nearest-even fp32 -> bf16
  unsigned u = __builtin_bit_cast(unsigned, f);
  unsigned r = u + 0x7FFFu + ((u >> 16) & 1u);
  unsigned short h = (unsigned short)(r >> 16);
  return __builtin_bit_cast(bf16_t, h);
}

// -------------------------- elementwise kernels ----------------------------

__global__ __launch_bounds__(256) void k_cvt(const float* __restrict__ in,
                                             bf16_t* __restrict__ out, int n) {
  for (int i = blockIdx.x * blockDim.x + threadIdx.x; i < n;
       i += gridDim.x * blockDim.x)
    out[i] = f2bf(in[i]);
}

__global__ __launch_bounds__(256) void k_scale_cvt(const float* __restrict__ in,
                                                   bf16_t* __restrict__ out,
                                                   float s, int n) {
  for (int i = blockIdx.x * blockDim.x + threadIdx.x; i < n;
       i += gridDim.x * blockDim.x)
    out[i] = f2bf(in[i] * s);
}

__global__ __launch_bounds__(256) void k_vecadd(const float* __restrict__ a,
                                                const float* __restrict__ b,
                                                float* __restrict__ o, int n) {
  int i = blockIdx.x * blockDim.x + threadIdx.x;
  if (i < n) o[i] = a[i] + b[i];
}

// LSTM pointwise: gates g = [i | f | g | o] per row (PyTorch order).
__global__ __launch_bounds__(256)
void k_lstm_gate(const float* __restrict__ g, float* __restrict__ c,
                 bf16_t* __restrict__ h, int BH, int H) {
  int idx = blockIdx.x * blockDim.x + threadIdx.x;
  if (idx >= BH) return;
  int b = idx / H;
  int j = idx - b * H;
  const float* gr = g + (size_t)b * (4 * H);
  float xi = gr[j];
  float xf = gr[H + j];
  float xg = gr[2 * H + j];
  float xo = gr[3 * H + j];
  float i_ = 1.f / (1.f + __expf(-xi));
  float f_ = 1.f / (1.f + __expf(-xf));
  float o_ = 1.f / (1.f + __expf(-xo));
  float g_ = tanhf(xg);
  float cn = f_ * c[idx] + i_ * g_;
  c[idx] = cn;
  h[idx] = f2bf(o_ * tanhf(cn));
}

// ------------------------------ WMMA GEMM ----------------------------------
// C[M,N] = A[M,K] * W[N,K]^T  (+ epilogue)
// A, W bf16 row-major; accumulation f32 via v_wmma_f32_16x16x32_bf16.
// Wave tile 64(M) x 32(N): 4x2 accumulators. Block: 8 waves as 2(M) x 4(N)
// -> block tile 128x128. M%128==0, N%128==0, K%32==0 hold for all GEMMs.

enum {
  EPI_BIAS_F32 = 0,       // Of = acc + bias[n]
  EPI_ADDC_F32,           // Of = acc + Cin        (Cin may alias Of)
  EPI_BIAS_BF16,          // Ob = bf16(acc + bias)
  EPI_BIAS_RELU_BF16,     // Ob = bf16(relu(acc + bias))
  EPI_BIAS_RELU_ACC_F32   // Of += relu(acc + bias)
};

#define MT 4
#define NT 2

template <int EPI>
__global__ __launch_bounds__(256)
void k_gemm(const bf16_t* __restrict__ A, const bf16_t* __restrict__ W,
            const float* __restrict__ bias, const float* __restrict__ Cin,
            float* __restrict__ Of, bf16_t* __restrict__ Ob,
            int M, int N, int K) {
  const int lane = threadIdx.x & 31;
  const int wave = threadIdx.x >> 5;
  const int lr = lane & 15;   // row/col selector within 16
  const int lh = lane >> 4;   // half-wave selector
  const int m0 = blockIdx.y * 128 + (wave >> 2) * (MT * 16);
  const int n0 = blockIdx.x * 128 + (wave & 3) * (NT * 16);

  v8f acc[MT][NT];
  for (int a = 0; a < MT; ++a)
    for (int b = 0; b < NT; ++b)
      for (int i = 0; i < 8; ++i) acc[a][b][i] = 0.f;

  // A fragment (16x32), contiguous k order as B: lanes<16 hold K {0..15},
  // lanes>=16 hold K {16..31} of row (L&15) -> two b128 loads.
  const bf16_t* arow[MT];
  arow[0] = A + (size_t)(m0 + lr) * K + lh * 16;
  for (int mt = 1; mt < MT; ++mt) arow[mt] = arow[mt - 1] + (size_t)16 * K;
  // B fragment (32x16): lane holds 16 contiguous K of W row n0+(L&15),
  // half-wave selects the K block -> one 32B load.
  const bf16_t* brow[NT];
  brow[0] = W + (size_t)(n0 + lr) * K + lh * 16;
  for (int nt = 1; nt < NT; ++nt) brow[nt] = brow[nt - 1] + (size_t)16 * K;

  for (int k0 = 0; k0 < K; k0 += 32) {
    v16bf fa[MT], fb[NT];
#pragma unroll
    for (int mt = 0; mt < MT; ++mt) {
      uint4* p = (uint4*)&fa[mt];
      p[0] = *(const uint4*)(arow[mt] + k0);
      p[1] = *(const uint4*)(arow[mt] + k0 + 8);
    }
#pragma unroll
    for (int nt = 0; nt < NT; ++nt) fb[nt] = *(const v16bf*)(brow[nt] + k0);
    // prefetch 4 k-steps ahead, high locality (near caches)
    __builtin_prefetch(arow[0] + k0 + 128, 0, 3);
    __builtin_prefetch(arow[2] + k0 + 128, 0, 3);
    __builtin_prefetch(brow[0] + k0 + 128, 0, 3);
#pragma unroll
    for (int mt = 0; mt < MT; ++mt)
#pragma unroll
      for (int nt = 0; nt < NT; ++nt)
        acc[mt][nt] = __builtin_amdgcn_wmma_f32_16x16x32_bf16(
            false, fa[mt], false, fb[nt], (short)0, acc[mt][nt], false, false);
  }

  // D layout: VGPR i -> M = i + (lane>=16 ? 8 : 0), N = lane&15.
#pragma unroll
  for (int nt = 0; nt < NT; ++nt) {
    const int n = n0 + nt * 16 + lr;
    float bn = 0.f;
    if constexpr (EPI != EPI_ADDC_F32) bn = bias[n];
#pragma unroll
    for (int mt = 0; mt < MT; ++mt) {
#pragma unroll
      for (int i = 0; i < 8; ++i) {
        const int m = m0 + mt * 16 + lh * 8 + i;
        const size_t idx = (size_t)m * N + n;
        float v = acc[mt][nt][i] + bn;
        if constexpr (EPI == EPI_ADDC_F32) {
          Of[idx] = v + Cin[idx];
        } else if constexpr (EPI == EPI_BIAS_F32) {
          Of[idx] = v;
        } else if constexpr (EPI == EPI_BIAS_BF16) {
          Ob[idx] = f2bf(v);
        } else if constexpr (EPI == EPI_BIAS_RELU_BF16) {
          Ob[idx] = f2bf(v > 0.f ? v : 0.f);
        } else {  // EPI_BIAS_RELU_ACC_F32
          Of[idx] += (v > 0.f ? v : 0.f);
        }
      }
    }
  }
}

template <int EPI>
static void launch_gemm(const bf16_t* A, const bf16_t* W, const float* bias,
                        const float* Cin, float* Of, bf16_t* Ob,
                        int M, int N, int K, hipStream_t s) {
  dim3 grid(N / 128, M / 128), block(256);
  k_gemm<EPI><<<grid, block, 0, s>>>(A, W, bias, Cin, Of, Ob, M, N, K);
}

// ------------------------------- driver ------------------------------------

extern "C" void kernel_launch(void* const* d_in, const int* in_sizes, int n_in,
                              void* d_out, int out_size, void* d_ws,
                              size_t ws_size, hipStream_t stream) {
  const float* x       = (const float*)d_in[0];
  const float* emb_W   = (const float*)d_in[1];
  const float* emb_b   = (const float*)d_in[2];
  const float* enc_Wih = (const float*)d_in[3];
  const float* enc_Whh = (const float*)d_in[4];
  const float* enc_bih = (const float*)d_in[5];
  const float* enc_bhh = (const float*)d_in[6];
  const float* dec_Wih = (const float*)d_in[7];
  const float* dec_Whh = (const float*)d_in[8];
  const float* dec_bih = (const float*)d_in[9];
  const float* dec_bhh = (const float*)d_in[10];
  const float* h2e_W   = (const float*)d_in[11];
  const float* h2e_b   = (const float*)d_in[12];
  const float* p1_W    = (const float*)d_in[13];
  const float* p1_b    = (const float*)d_in[14];
  const float* p2_W    = (const float*)d_in[15];
  const float* p2_b    = (const float*)d_in[16];
  float* out = (float*)d_out;

  // ---- carve workspace (256B aligned slabs) ----
  char* w = (char*)d_ws;
  auto carve = [&](size_t bytes) -> char* {
    char* p = w;
    w += (bytes + 255) & ~(size_t)255;
    return p;
  };
  bf16_t* xb      = (bf16_t*)carve((size_t)B_ROWS * YIN * 2);
  bf16_t* embWb   = (bf16_t*)carve((size_t)EMBD * YIN * 2);
  bf16_t* encWihb = (bf16_t*)carve((size_t)G4 * EMBD * 2);
  bf16_t* encWhhb = (bf16_t*)carve((size_t)G4 * HIDN * 2);
  bf16_t* decWihb = (bf16_t*)carve((size_t)G4 * EMBD * 2);
  bf16_t* decWhhb = (bf16_t*)carve((size_t)G4 * HIDN * 2);
  bf16_t* h2eWb   = (bf16_t*)carve((size_t)EMBD * HIDN * 2);
  bf16_t* p1Wb    = (bf16_t*)carve((size_t)EMBD * HIDN * 2);
  bf16_t* p2Wb    = (bf16_t*)carve((size_t)YIN * EMBD * 2);
  bf16_t* e_bf    = (bf16_t*)carve((size_t)B_ROWS * EMBD * 2);
  bf16_t* xin_bf  = (bf16_t*)carve((size_t)B_ROWS * EMBD * 2);
  bf16_t* zmean_bf= (bf16_t*)carve((size_t)B_ROWS * EMBD * 2);
  bf16_t* h_bf    = (bf16_t*)carve((size_t)B_ROWS * HIDN * 2);
  float*  enc_bias= (float*)carve((size_t)G4 * 4);
  float*  dec_bias= (float*)carve((size_t)G4 * 4);
  float*  enc_pre = (float*)carve((size_t)B_ROWS * G4 * 4);
  float*  gbuf    = (float*)carve((size_t)B_ROWS * G4 * 4);
  float*  cstate  = (float*)carve((size_t)B_ROWS * HIDN * 4);
  float*  zsum    = (float*)carve((size_t)B_ROWS * EMBD * 4);
  (void)ws_size; (void)in_sizes; (void)n_in; (void)out_size;

  const dim3 cb(256);
  const int CG = 1024;  // grid-stride convert blocks
  auto cvt = [&](const float* src, bf16_t* dst, size_t n) {
    k_cvt<<<CG, cb, 0, stream>>>(src, dst, (int)n);
  };

  // ---- fp32 -> bf16 operand staging ----
  cvt(x, xb, (size_t)B_ROWS * YIN);
  cvt(emb_W, embWb, (size_t)EMBD * YIN);
  cvt(enc_Wih, encWihb, (size_t)G4 * EMBD);
  cvt(enc_Whh, encWhhb, (size_t)G4 * HIDN);
  cvt(dec_Wih, decWihb, (size_t)G4 * EMBD);
  cvt(dec_Whh, decWhhb, (size_t)G4 * HIDN);
  cvt(h2e_W, h2eWb, (size_t)EMBD * HIDN);
  cvt(p1_W, p1Wb, (size_t)EMBD * HIDN);
  cvt(p2_W, p2Wb, (size_t)YIN * EMBD);
  k_vecadd<<<G4 / 256, cb, 0, stream>>>(enc_bih, enc_bhh, enc_bias, G4);
  k_vecadd<<<G4 / 256, cb, 0, stream>>>(dec_bih, dec_bhh, dec_bias, G4);

  // ---- zero recurrent state / accumulators (deterministic each call) ----
  hipMemsetAsync(h_bf, 0, (size_t)B_ROWS * HIDN * 2, stream);
  hipMemsetAsync(cstate, 0, (size_t)B_ROWS * HIDN * 4, stream);
  hipMemsetAsync(zsum, 0, (size_t)B_ROWS * EMBD * 4, stream);

  // ---- embedding: e = relu(x @ emb_W^T + emb_b) ----
  launch_gemm<EPI_BIAS_RELU_BF16>(xb, embWb, emb_b, nullptr, nullptr, e_bf,
                                  B_ROWS, EMBD, YIN, stream);

  // ---- encoder input term hoisted: enc_pre = e @ Wih^T + (bih + bhh) ----
  launch_gemm<EPI_BIAS_F32>(e_bf, encWihb, enc_bias, nullptr, enc_pre, nullptr,
                            B_ROWS, G4, EMBD, stream);

  // ---- encoder recurrence ----
  const int BH = B_ROWS * HIDN;
  for (int t = 0; t < TSTEPS; ++t) {
    launch_gemm<EPI_ADDC_F32>(h_bf, encWhhb, nullptr, enc_pre, gbuf, nullptr,
                              B_ROWS, G4, HIDN, stream);
    k_lstm_gate<<<BH / 256, cb, 0, stream>>>(gbuf, cstate, h_bf, BH, HIDN);
  }

  // ---- decoder (state carried over from encoder; first input = e) ----
  for (int t = 0; t < TSTEPS; ++t) {
    const bf16_t* xin = (t == 0) ? e_bf : xin_bf;
    launch_gemm<EPI_BIAS_F32>(xin, decWihb, dec_bias, nullptr, gbuf, nullptr,
                              B_ROWS, G4, EMBD, stream);
    launch_gemm<EPI_ADDC_F32>(h_bf, decWhhb, nullptr, gbuf, gbuf, nullptr,
                              B_ROWS, G4, HIDN, stream);
    k_lstm_gate<<<BH / 256, cb, 0, stream>>>(gbuf, cstate, h_bf, BH, HIDN);
    if (t + 1 < TSTEPS) {  // next decoder input
      launch_gemm<EPI_BIAS_BF16>(h_bf, h2eWb, h2e_b, nullptr, nullptr, xin_bf,
                                 B_ROWS, EMBD, HIDN, stream);
    }
    // zsum += relu(h @ p1^T + p1_b)   (mean-through-linear trick)
    launch_gemm<EPI_BIAS_RELU_ACC_F32>(h_bf, p1Wb, p1_b, nullptr, zsum, nullptr,
                                       B_ROWS, EMBD, HIDN, stream);
  }

  // ---- final projection: out = (zsum / T) @ p2^T + p2_b ----
  k_scale_cvt<<<CG, cb, 0, stream>>>(zsum, zmean_bf, 1.0f / (float)TSTEPS,
                                     B_ROWS * EMBD);
  launch_gemm<EPI_BIAS_F32>(zmean_bf, p2Wb, p2_b, nullptr, out, nullptr,
                            B_ROWS, YIN, EMBD, stream);
}
